// LSTM_29987461661200
// MI455X (gfx1250) — matrix operand order repaired and checked by reading, not compiled, not run on hardware
//
#include <hip/hip_runtime.h>
#include <hip/hip_bf16.h>

// ---------------------------------------------------------------------------
// LSTM on MI455X (gfx1250), bf16 WMMA with fp32 accumulation.
//   Phase 0a: pack W_g / U_g (fp32 [K=512][N=512]) into WMMA-B bf16 lane layout
//   Phase 0b: pack x (fp32 [B,T,D]) into bf16 time-major [t*64+b][k] rows
//   Phase 1 : XprojPk[g][t][tile][lane][r] = (x @ W_g + b_g) in C-operand layout
//   Phase 2 : persistent-grid recurrence; U operands staged once into 256 KB
//             of LDS per WGP, then 512 steps of pure LDS-fed WMMA with a
//             device-wide barrier per step.
// ---------------------------------------------------------------------------

typedef __attribute__((ext_vector_type(16))) __bf16 v16bf;
typedef __attribute__((ext_vector_type(8)))  float v8f;

union V16U {
    v16bf v;
    unsigned short u[16];
    uint4 q[2];
};

__device__ __forceinline__ unsigned short f2bfu(float f) {
    union { float f; unsigned u; } x; x.f = f;
    unsigned r = x.u + 0x7FFFu + ((x.u >> 16) & 1u);   // round-to-nearest-even
    return (unsigned short)(r >> 16);
}

__device__ __forceinline__ v8f wmma_bf16(v16bf a, v16bf b, v8f c) {
    // D(f32 16x16) = A(bf16 16x32) * B(bf16 32x16) + C(f32)
    return __builtin_amdgcn_wmma_f32_16x16x32_bf16(
        /*neg_a=*/false, a, /*neg_b=*/false, b,
        /*c_mod=*/(short)0, c, /*reuse_a=*/false, /*reuse_b=*/false);
}

__device__ __forceinline__ float sigmoidf_(float x) {
    return 1.0f / (1.0f + __expf(-x));
}

// tanh(x) = 1 - 2/(e^{2x}+1): one transcendental + one rcp, saturates
// correctly to +/-1 for large |x| (no inf/inf NaN path).
__device__ __forceinline__ float tanhf_(float x) {
    return 1.0f - 2.0f / (__expf(2.0f * x) + 1.0f);
}

// ---------------------------------------------------------------------------
// Pack one 512x512 fp32 matrix (row-major [k][n]) into WMMA-B bf16 layout:
//   dst[((kc*32 + nt)*32 + lane)*16 + e] = src[kc*32 + 16*(lane>>4) + e][nt*16 + (lane&15)]
// so each lane's 16 operand bf16s are 32 contiguous bytes (two b128 loads).
// ---------------------------------------------------------------------------
__global__ void lstm_pack_b(const float* __restrict__ src,
                            unsigned short* __restrict__ dst) {
    for (int idx = blockIdx.x * blockDim.x + threadIdx.x; idx < 512 * 512;
         idx += gridDim.x * blockDim.x) {
        const int e    = idx & 15;
        const int lane = (idx >> 4) & 31;
        const int nt   = (idx >> 9) & 31;
        const int kc   = idx >> 14;
        const int k = kc * 32 + ((lane >> 4) << 4) + e;
        const int n = nt * 16 + (lane & 15);
        dst[idx] = f2bfu(src[(size_t)k * 512 + n]);
    }
}

// ---------------------------------------------------------------------------
// Pack x [B=64][T=512][D=512] fp32 into bf16 time-major rows:
//   xbf[(t*64 + b)*512 + k] = bf16(x[b][t][k])
// ---------------------------------------------------------------------------
__global__ void lstm_pack_x(const float* __restrict__ x,
                            unsigned short* __restrict__ xbf) {
    const int total = 32768 * 128;   // float4 granules
    for (int i = blockIdx.x * blockDim.x + threadIdx.x; i < total;
         i += gridDim.x * blockDim.x) {
        const int k4 = i & 127;
        const int m  = i >> 7;          // m = t*64 + b
        const int b  = m & 63;
        const int t  = m >> 6;
        const float4 v = ((const float4*)(x + ((size_t)(b * 512 + t)) * 512))[k4];
        union { unsigned short s[4]; unsigned long long ll; } p;
        p.s[0] = f2bfu(v.x);
        p.s[1] = f2bfu(v.y);
        p.s[2] = f2bfu(v.z);
        p.s[3] = f2bfu(v.w);
        *(unsigned long long*)(xbf + (size_t)m * 512 + k4 * 4) = p.ll;
    }
}

// ---------------------------------------------------------------------------
// Phase 1: XprojPk in WMMA C-operand lane layout:
//   XprojPk[(((g*512 + t)*128 + tile2)*32 + lane)*8 + r]
// tile2 = (mt&3)*32 + nt matches the recurrence tile grid; t = mt>>2.
// One wave per 16x16 tile, 4 gates sharing A.
// ---------------------------------------------------------------------------
__global__ void lstm_xproj(const unsigned short* __restrict__ xbf,
                           const unsigned short* __restrict__ pkW,
                           const float* __restrict__ b_c, const float* __restrict__ b_i,
                           const float* __restrict__ b_f, const float* __restrict__ b_o,
                           float* __restrict__ xprojpk) {
    const int lane  = threadIdx.x & 31;
    const int wv    = threadIdx.x >> 5;
    const int gw    = blockIdx.x * (blockDim.x >> 5) + wv;
    const int nw    = gridDim.x * (blockDim.x >> 5);
    const int khalf = lane >> 4;
    const int lnib  = lane & 15;

    for (int tile = gw; tile < 2048 * 32; tile += nw) {
        const int mt = tile >> 5;
        const int nt = tile & 31;
        const int n  = nt * 16 + lnib;
        const int m  = mt * 16 + lnib;          // A row handled by this lane
        const unsigned short* arow = xbf + (size_t)m * 512 + 16 * khalf;
        const unsigned short* brow = pkW + ((size_t)nt * 32 + lane) * 16;

        v8f acc[4];
        #pragma unroll
        for (int g = 0; g < 4; ++g)
            #pragma unroll
            for (int r = 0; r < 8; ++r) acc[g][r] = 0.0f;

        #pragma unroll 4
        for (int kc = 0; kc < 16; ++kc) {
            const unsigned short* ap = arow + kc * 32;
            V16U au;
            au.q[0] = *(const uint4*)(ap);
            au.q[1] = *(const uint4*)(ap + 8);
            #pragma unroll
            for (int g = 0; g < 4; ++g) {
                const unsigned short* bp =
                    brow + (size_t)g * 262144 + (size_t)kc * 16384;
                V16U bu;
                bu.q[0] = ((const uint4*)bp)[0];
                bu.q[1] = ((const uint4*)bp)[1];
                acc[g] = wmma_bf16(au.v, bu.v, acc[g]);
            }
        }

        const float bias[4] = { b_c[n], b_i[n], b_f[n], b_o[n] };
        const int t     = mt >> 2;
        const int tile2 = (mt & 3) * 32 + nt;
        #pragma unroll
        for (int g = 0; g < 4; ++g) {
            v8f vout;
            #pragma unroll
            for (int r = 0; r < 8; ++r) vout[r] = acc[g][r] + bias[g];
            float* dst = xprojpk +
                ((((size_t)g * 512 + t) * 128 + tile2) * 32 + lane) * 8;
            *(v8f*)dst = vout;      // 32B contiguous per lane -> two b128 stores
        }
    }
}

// ---------------------------------------------------------------------------
// Phase 2: persistent recurrence.
//   8 WGs x 8 waves = 64 waves.  Wave wv of WG b: ntl = wv&3, mh = wv>>2,
//   nt = b*4 + ntl.  Each wave computes a 32x16 output slab (m-tiles 2mh and
//   2mh+1) for its nt, all 4 gates.  U operands for the WG's 4 nt-columns
//   (4 x 64 KB) are staged into LDS once; the 512-step loop then feeds all
//   WMMAs from ds_load_b128.  Cell state in registers; h double-buffered
//   bf16 in global; device-wide barrier (8 WGs) per step.
// ---------------------------------------------------------------------------
__global__ void lstm_recur(const unsigned short* __restrict__ pkU,
                           const float* __restrict__ xprojpk,
                           unsigned short* __restrict__ hbuf,   // [2][64*512] bf16
                           unsigned int* __restrict__ counter,
                           float* __restrict__ out) {            // [64][512][512]
    __shared__ unsigned short bstage[131072];  // 256 KB: [ntl][g*16+kc][lane*16+e]

    const int lane  = threadIdx.x & 31;
    const int wv    = threadIdx.x >> 5;         // 0..7
    const int ntl   = wv & 3;                   // local nt column
    const int mh    = wv >> 2;                  // m-half: rows [mh*32, mh*32+32)
    const int nt    = blockIdx.x * 4 + ntl;     // 0..31
    const int khalf = lane >> 4;
    const int lnib  = lane & 15;
    const int n     = nt * 16 + lnib;
    const unsigned int nwg = gridDim.x;

    // ---- one-time: stage U operands for this WG's 4 nt-columns into LDS ----
    // Column ntl (64 KB) is filled by the two waves sharing it (mh=0: first
    // 32 (g,kc) pairs, mh=1: last 32).
    {
        const unsigned short* brow = pkU + ((size_t)nt * 32 + lane) * 16;
        const int base = ntl * 32768;
        for (int j = 0; j < 32; ++j) {
            const int i = mh * 32 + j;          // i = g*16 + kc
            const unsigned short* bp = brow + (size_t)(i >> 4) * 262144
                                            + (size_t)(i & 15) * 16384;
            uint4 q0 = ((const uint4*)bp)[0];
            uint4 q1 = ((const uint4*)bp)[1];
            const int d = base + i * 512 + lane * 16;
            *(uint4*)&bstage[d]     = q0;
            *(uint4*)&bstage[d + 8] = q1;
        }
    }
    __syncthreads();

    const int bsbase = ntl * 32768;
    const int tile0  = (mh * 2) * 32 + nt;      // 128-tile-grid ids of the
    const int tile1  = (mh * 2 + 1) * 32 + nt;  // two m-tiles this wave owns

    float creg[2][8];
    #pragma unroll
    for (int p = 0; p < 2; ++p)
        #pragma unroll
        for (int r = 0; r < 8; ++r) creg[p][r] = 0.0f;

    for (int t = 0; t < 512; ++t) {
        // init accumulators with precomputed x-projections (C-operand layout)
        v8f acc[2][4];
        #pragma unroll
        for (int g = 0; g < 4; ++g) {
            acc[0][g] = *(const v8f*)(xprojpk +
                ((((size_t)g * 512 + t) * 128 + tile0) * 32 + lane) * 8);
            acc[1][g] = *(const v8f*)(xprojpk +
                ((((size_t)g * 512 + t) * 128 + tile1) * 32 + lane) * 8);
        }

        if (t > 0) {  // h_0 == 0: skip the GEMM entirely on the first step
            const unsigned short* hp0 = hbuf + (size_t)(t & 1) * 32768
                + (size_t)(mh * 32 + lnib) * 512 + 16 * khalf;   // m-tile 2mh
            const unsigned short* hp1 = hp0 + 16 * 512;          // m-tile 2mh+1
            #pragma unroll
            for (int kc = 0; kc < 16; ++kc) {
                V16U a0, a1;
                a0.q[0] = *(const uint4*)(hp0 + kc * 32);
                a0.q[1] = *(const uint4*)(hp0 + kc * 32 + 8);
                a1.q[0] = *(const uint4*)(hp1 + kc * 32);
                a1.q[1] = *(const uint4*)(hp1 + kc * 32 + 8);
                #pragma unroll
                for (int g = 0; g < 4; ++g) {
                    const int bd = bsbase + (g * 16 + kc) * 512 + lane * 16;
                    V16U bu;                      // LDS: two ds_load_b128
                    bu.q[0] = *(const uint4*)&bstage[bd];
                    bu.q[1] = *(const uint4*)&bstage[bd + 8];
                    acc[0][g] = wmma_bf16(a0.v, bu.v, acc[0][g]);
                    acc[1][g] = wmma_bf16(a1.v, bu.v, acc[1][g]);
                }
            }
        }

        // elementwise gates + state update + stores (both m-tiles)
        unsigned short* hdst = hbuf + (size_t)((t + 1) & 1) * 32768;
        #pragma unroll
        for (int p = 0; p < 2; ++p) {
            #pragma unroll
            for (int r = 0; r < 8; ++r) {
                const int m = mh * 32 + p * 16 + 8 * khalf + r;   // batch index
                const float av = tanhf_(acc[p][0][r]);
                const float iv = sigmoidf_(acc[p][1][r]);
                const float fv = sigmoidf_(acc[p][2][r]);
                const float ov = sigmoidf_(acc[p][3][r]);
                const float cn = iv * av + fv * creg[p][r];
                creg[p][r] = cn;
                const float hn = ov * tanhf_(cn);
                out[((size_t)m * 512 + t) * 512 + n] = hn;   // out[b][t][h]
                hdst[(size_t)m * 512 + n] = f2bfu(hn);
            }
        }

        // ---- device-wide barrier: all h_new visible before next step ----
        __threadfence();            // release our h stores to device scope
        __syncthreads();
        if (threadIdx.x == 0) {
            __hip_atomic_fetch_add(counter, 1u, __ATOMIC_RELEASE,
                                   __HIP_MEMORY_SCOPE_AGENT);
            const unsigned int target = nwg * (unsigned int)(t + 1);
            while (__hip_atomic_load(counter, __ATOMIC_ACQUIRE,
                                     __HIP_MEMORY_SCOPE_AGENT) < target) {
                __builtin_amdgcn_s_sleep(2);
            }
        }
        __syncthreads();
        __threadfence();            // acquire: drop stale h lines from WGP$
    }
}

// ---------------------------------------------------------------------------
// Host side. Inputs (dict order): x, W_c, W_i, W_f, W_o, U_c, U_i, U_f, U_o,
// b_c, b_i, b_f, b_o.  All fp32. Output: [64][512][512] fp32.
// Workspace layout (bytes):
//   [0,256)            barrier counter
//   [256, +2MB)        pkW   (4 x 512x512 bf16)
//   [.., +2MB)         pkU   (4 x 512x512 bf16)
//   [.., +128KB)       hbuf  (2 x 64x512 bf16)
//   [.., +32MB)        xbf   (32768 x 512 bf16, time-major)
//   [.., +256MB)       xprojpk (4 x 512 x 128 x 32 x 8 fp32)
// ---------------------------------------------------------------------------
extern "C" void kernel_launch(void* const* d_in, const int* in_sizes, int n_in,
                              void* d_out, int out_size, void* d_ws, size_t ws_size,
                              hipStream_t stream) {
    const float* x = (const float*)d_in[0];
    const float* W[4] = { (const float*)d_in[1], (const float*)d_in[2],
                          (const float*)d_in[3], (const float*)d_in[4] };
    const float* U[4] = { (const float*)d_in[5], (const float*)d_in[6],
                          (const float*)d_in[7], (const float*)d_in[8] };
    const float* bias[4] = { (const float*)d_in[9],  (const float*)d_in[10],
                             (const float*)d_in[11], (const float*)d_in[12] };
    float* out = (float*)d_out;

    char* ws = (char*)d_ws;
    unsigned int*   counter = (unsigned int*)ws;
    unsigned short* pkW     = (unsigned short*)(ws + 256);
    unsigned short* pkU     = pkW + (size_t)4 * 262144;
    unsigned short* hbuf    = pkU + (size_t)4 * 262144;
    unsigned short* xbf     = hbuf + (size_t)2 * 64 * 512;
    float*          xprojpk = (float*)(ws + 256
                                       + (size_t)8 * 262144 * 2      // pkW + pkU
                                       + (size_t)2 * 64 * 512 * 2    // hbuf
                                       + (size_t)32768 * 512 * 2);   // xbf

    hipMemsetAsync(counter, 0, 256, stream);

    for (int g = 0; g < 4; ++g)
        lstm_pack_b<<<256, 256, 0, stream>>>(W[g], pkW + (size_t)g * 262144);
    for (int g = 0; g < 4; ++g)
        lstm_pack_b<<<256, 256, 0, stream>>>(U[g], pkU + (size_t)g * 262144);
    lstm_pack_x<<<4096, 256, 0, stream>>>(x, xbf);

    lstm_xproj<<<1024, 256, 0, stream>>>(xbf, pkW, bias[0], bias[1], bias[2],
                                         bias[3], xprojpk);

    lstm_recur<<<8, 256, 0, stream>>>(pkU, xprojpk, hbuf, counter, out);
}
